// base_model_81381040325052
// MI455X (gfx1250) — matrix operand order repaired and checked by reading, not compiled, not run on hardware
//
#include <hip/hip_runtime.h>
#include <hip/hip_bf16.h>

#define N_NODES   20000
#define N_EDGES   200000
#define N_GRAPHS  128
#define EMB       256
#define N_LAYERS  4
#define RBF_DIM   10
#define RBF_CUTOFF 8.0f

typedef _Float16 v16h __attribute__((ext_vector_type(16)));
typedef _Float16 v8h  __attribute__((ext_vector_type(8)));
typedef float    v8f  __attribute__((ext_vector_type(8)));

#define AROW 264   // 256 + 8 halves padding -> rows shift 4 banks, keeps 16B alignment

// ---------------- WMMA helpers ----------------

__device__ __forceinline__ v8f wmma_f16(v16h a, v16h b, v8f c) {
    // v_wmma_f32_16x16x32_f16 : D(16x16 f32) = A(16x32 f16) x B(32x16 f16) + C
    return __builtin_amdgcn_wmma_f32_16x16x32_f16(
        /*neg_a=*/false, a, /*neg_b=*/false, b,
        /*c_mod=*/(short)0, c, /*reuse_a=*/false, /*reuse_b=*/false);
}

// A fragment from LDS tile (row-major, stride AROW halves).
// ISA layout (16-bit A 16x32): lane L: M=L%16, halves = K[kb..kb+7], K[kb+16..kb+23], kb=(L>=16)?8:0
__device__ __forceinline__ v16h load_a_frag(const _Float16* Ash, int mt, int kt, int lane) {
    int r  = lane & 15;
    int hi = lane >> 4;
    const _Float16* base = Ash + (mt * 16 + r) * AROW + kt * 32 + hi * 8;
    v8h lo = *(const v8h*)base;
    v8h hh = *(const v8h*)(base + 16);
    v16h a;
#pragma unroll
    for (int i = 0; i < 8; ++i) { a[i] = lo[i]; a[i + 8] = hh[i]; }
    return a;
}

// B fragment from global transposed weights Wt (f16, layout [n][k], row stride 256).
// ISA layout (16-bit B 32x16): lane L: N=L%16, halves = K[kb2..kb2+7], K[kb2+16..kb2+23], kb2=(L>=16)?8:0
__device__ __forceinline__ v16h load_b_frag(const _Float16* __restrict__ Wt, int nt, int kt, int lane) {
    int col = lane & 15;
    int kb  = (lane >> 4) * 8;
    const _Float16* base = Wt + (size_t)(nt * 16 + col) * 256 + kt * 32 + kb;
    v8h b0 = *(const v8h*)base;
    v8h b1 = *(const v8h*)(base + 16);
    v16h b;
#pragma unroll
    for (int i = 0; i < 8; ++i) { b[i] = b0[i]; b[i + 8] = b1[i]; }
    return b;
}

// ---------------- small kernels ----------------

// Transpose + f32->f16 all 6 weight tensors x 4 layers: out[mat][l][n][k] = W[mat][l][k][n]
__global__ __launch_bounds__(256) void convert_weights_kernel(
    const float* __restrict__ Wgx, const float* __restrict__ Wgn, const float* __restrict__ Wge,
    const float* __restrict__ Wfx, const float* __restrict__ Wfn, const float* __restrict__ Wfe,
    _Float16* __restrict__ out)
{
    int t = blockIdx.x * 256 + threadIdx.x;
    if (t >= 6 * N_LAYERS * EMB * EMB) return;
    int mat = t >> 18;            // 4*65536
    int rem = t & 262143;
    int l   = rem >> 16;
    int idx = rem & 65535;
    int n = idx >> 8, k = idx & 255;
    const float* W =
        (mat == 0) ? Wgx : (mat == 1) ? Wgn : (mat == 2) ? Wge :
        (mat == 3) ? Wfx : (mat == 4) ? Wfn : Wfe;
    out[t] = (_Float16)W[(size_t)l * 65536 + (size_t)k * 256 + n];
}

__global__ __launch_bounds__(256) void embed_kernel(
    const int* __restrict__ x, const float* __restrict__ emb, float* __restrict__ xn)
{
    int t = blockIdx.x * 256 + threadIdx.x;
    if (t >= N_NODES * EMB) return;
    int n = t >> 8, c = t & 255;
    xn[t] = emb[(size_t)x[n] * EMB + c];
}

__global__ __launch_bounds__(256) void rbf_kernel(
    const float* __restrict__ e, float* __restrict__ rbf)
{
    int t = blockIdx.x * 256 + threadIdx.x;
    if (t >= N_EDGES * RBF_DIM) return;
    int ei = t / RBF_DIM, k = t % RBF_DIM;
    float width = RBF_CUTOFF / (RBF_DIM - 1);
    float u = (e[ei] - (float)k * width) / width;
    rbf[t] = __expf(-u * u);
}

// ef16[e][c] = f16( rbf[e] @ We[:,c] + be[c] )
__global__ __launch_bounds__(256) void ef_kernel(
    const float* __restrict__ rbf, const float* __restrict__ We, const float* __restrict__ be,
    _Float16* __restrict__ ef16)
{
    int t = blockIdx.x * 256 + threadIdx.x;
    if (t >= N_EDGES * EMB) return;
    int e = t >> 8, c = t & 255;
    float acc = be[c];
#pragma unroll
    for (int k = 0; k < RBF_DIM; ++k)
        acc += rbf[e * RBF_DIM + k] * We[k * EMB + c];
    ef16[t] = (_Float16)acc;
}

// ---------------- node transform: 4 GEMMs sharing A = xn ----------------
// Block: 256 threads = 8 waves; tile 64 rows x 256 cols; wave w owns ntiles {2w, 2w+1}.
// Matrices processed sequentially so the resident B-fragment set is bounded:
// per matrix, bfr[8 kt][2 nt] = 128 VGPRs, reused by all 4 row-tiles.
__global__ __launch_bounds__(256) void node_transform_kernel(
    const float* __restrict__ X,
    const _Float16* __restrict__ Wt0, const _Float16* __restrict__ Wt1,
    const _Float16* __restrict__ Wt2, const _Float16* __restrict__ Wt3,
    float* __restrict__ O0, float* __restrict__ O1,
    float* __restrict__ O2, float* __restrict__ O3, int M)
{
    __shared__ _Float16 Ash[64][AROW];
    int m0 = blockIdx.x * 64;
    // stage A tile (f32 -> f16)
    for (int it = threadIdx.x; it < 64 * 64; it += 256) {
        int r = it >> 6, q = it & 63;
        int gr = m0 + r;
        float4 v = make_float4(0.f, 0.f, 0.f, 0.f);
        if (gr < M) v = ((const float4*)X)[(size_t)gr * 64 + q];
        _Float16* p = &Ash[r][q * 4];
        p[0] = (_Float16)v.x; p[1] = (_Float16)v.y; p[2] = (_Float16)v.z; p[3] = (_Float16)v.w;
    }
    __syncthreads();

    int lane = threadIdx.x & 31;
    int wave = threadIdx.x >> 5;
    int nt0  = wave * 2;
    int col  = lane & 15, hi = lane >> 4;

#pragma unroll 1
    for (int m = 0; m < 4; ++m) {
        const _Float16* Wt = (m == 0) ? Wt0 : (m == 1) ? Wt1 : (m == 2) ? Wt2 : Wt3;
        float*          O  = (m == 0) ? O0  : (m == 1) ? O1  : (m == 2) ? O2  : O3;

        // preload this matrix's B fragments for our 2 ntiles, all ktiles
        v16h bfr[8][2];
#pragma unroll
        for (int kt = 0; kt < 8; ++kt) {
            bfr[kt][0] = load_b_frag(Wt, nt0 + 0, kt, lane);
            bfr[kt][1] = load_b_frag(Wt, nt0 + 1, kt, lane);
        }

#pragma unroll 1
        for (int mt = 0; mt < 4; ++mt) {
            v8f acc0, acc1;
#pragma unroll
            for (int g = 0; g < 8; ++g) { acc0[g] = 0.0f; acc1[g] = 0.0f; }
#pragma unroll
            for (int kt = 0; kt < 8; ++kt) {
                v16h a = load_a_frag(&Ash[0][0], mt, kt, lane);
                acc0 = wmma_f16(a, bfr[kt][0], acc0);
                acc1 = wmma_f16(a, bfr[kt][1], acc1);
            }
#pragma unroll
            for (int g = 0; g < 8; ++g) {
                int row = m0 + mt * 16 + hi * 8 + g;
                if (row < M) {
                    O[(size_t)row * EMB + (nt0 + 0) * 16 + col] = acc0[g];
                    O[(size_t)row * EMB + (nt0 + 1) * 16 + col] = acc1[g];
                }
            }
        }
    }
}

// ---------------- fused edge layer: cg/cf WMMA + gather + gate + scatter ----------------
// Block handles 64 edges x 256 channels. N_EDGES % 64 == 0.
// ntiles processed sequentially; per ntile the resident B set is 2 matrices x 8 kt = 128 VGPRs.
__global__ __launch_bounds__(256) void edge_layer_kernel(
    const _Float16* __restrict__ EF,
    const float* __restrict__ AG, const float* __restrict__ AN,
    const float* __restrict__ AF, const float* __restrict__ AFN,
    const _Float16* __restrict__ WgeT, const _Float16* __restrict__ WfeT,
    const float* __restrict__ bg, const float* __restrict__ bf,
    const int* __restrict__ src, const int* __restrict__ dst,
    float* __restrict__ Xout)
{
    __shared__ _Float16 Esh[64][AROW];
    __shared__ int s_src[64];
    __shared__ int s_dst[64];
    int e0 = blockIdx.x * 64;

    if (threadIdx.x < 64)       s_src[threadIdx.x]      = src[e0 + threadIdx.x];
    else if (threadIdx.x < 128) s_dst[threadIdx.x - 64] = dst[e0 + threadIdx.x - 64];

    // stage ef tile (already f16): 64 rows x 32 chunks of 8 halves
    for (int it = threadIdx.x; it < 64 * 32; it += 256) {
        int r = it >> 5, q = it & 31;
        uint4 v = *(const uint4*)(EF + (size_t)(e0 + r) * EMB + q * 8);
        *(uint4*)&Esh[r][q * 8] = v;
    }
    __syncthreads();

    int lane = threadIdx.x & 31;
    int wave = threadIdx.x >> 5;
    int nt0  = wave * 2;
    int col  = lane & 15, hi = lane >> 4;

#pragma unroll 1
    for (int n = 0; n < 2; ++n) {
        int nt = nt0 + n;
        int c  = nt * 16 + col;
        float bgc = bg[c];
        float bfc = bf[c];

        // preload B fragments for this ntile: both gate matrices, all ktiles
        v16h bg_fr[8], bf_fr[8];
#pragma unroll
        for (int kt = 0; kt < 8; ++kt) {
            bg_fr[kt] = load_b_frag(WgeT, nt, kt, lane);
            bf_fr[kt] = load_b_frag(WfeT, nt, kt, lane);
        }

#pragma unroll 1
        for (int mt = 0; mt < 4; ++mt) {
            v8f accg, accf;
#pragma unroll
            for (int g = 0; g < 8; ++g) { accg[g] = 0.0f; accf[g] = 0.0f; }
#pragma unroll
            for (int kt = 0; kt < 8; ++kt) {
                v16h a = load_a_frag(&Esh[0][0], mt, kt, lane);
                accg = wmma_f16(a, bg_fr[kt], accg);
                accf = wmma_f16(a, bf_fr[kt], accf);
            }
#pragma unroll
            for (int g = 0; g < 8; ++g) {
                int er = mt * 16 + hi * 8 + g;   // edge row within block
                int d  = s_dst[er];
                int s  = s_src[er];
                float zg = accg[g] + AG[(size_t)d * EMB + c] + AN[(size_t)s * EMB + c] + bgc;
                float zf = accf[g] + AF[(size_t)d * EMB + c] + AFN[(size_t)s * EMB + c] + bfc;
                float sig = 1.0f / (1.0f + __expf(-zg));
                float sp  = fmaxf(zf, 0.0f) + log1pf(__expf(-fabsf(zf)));
                atomicAdd(&Xout[(size_t)d * EMB + c], sig * sp);
            }
        }
    }
}

// ---------------- readout ----------------

__global__ __launch_bounds__(256) void zero_kernel(float* __restrict__ sums, float* __restrict__ cnt)
{
    int t = blockIdx.x * 256 + threadIdx.x;
    if (t < N_GRAPHS * 12) sums[t] = 0.0f;
    if (t < N_GRAPHS)      cnt[t]  = 0.0f;
}

__global__ __launch_bounds__(256) void readout_kernel(
    const float* __restrict__ X, const float* __restrict__ Wn, const float* __restrict__ bn,
    const int* __restrict__ gi, float* __restrict__ sums, float* __restrict__ cnt)
{
    int t = blockIdx.x * 256 + threadIdx.x;
    if (t >= N_NODES * 12) return;
    int n = t / 12, j = t % 12;
    float acc = bn[j];
    const float* xr = X + (size_t)n * EMB;
#pragma unroll 8
    for (int c = 0; c < EMB; ++c) acc += xr[c] * Wn[c * 12 + j];
    int g = gi[n];
    atomicAdd(&sums[g * 12 + j], acc);
    if (j == 0) atomicAdd(&cnt[g], 1.0f);
}

__global__ __launch_bounds__(256) void finalize_kernel(
    const float* __restrict__ sums, const float* __restrict__ cnt, float* __restrict__ out)
{
    int t = blockIdx.x * 256 + threadIdx.x;
    if (t >= N_GRAPHS * 12) return;
    out[t] = sums[t] / fmaxf(cnt[t / 12], 1.0f);
}

// ---------------- host orchestration ----------------

extern "C" void kernel_launch(void* const* d_in, const int* in_sizes, int n_in,
                              void* d_out, int out_size, void* d_ws, size_t ws_size,
                              hipStream_t stream) {
    (void)in_sizes; (void)n_in; (void)out_size; (void)ws_size;
    const int*   x    = (const int*)d_in[0];
    const int*   src  = (const int*)d_in[1];
    const int*   dst  = (const int*)d_in[2];
    const float* e    = (const float*)d_in[3];
    const int*   gi   = (const int*)d_in[4];
    const float* emb  = (const float*)d_in[5];
    const float* We   = (const float*)d_in[6];
    const float* be   = (const float*)d_in[7];
    const float* Wg_x = (const float*)d_in[8];
    const float* Wg_n = (const float*)d_in[9];
    const float* Wg_e = (const float*)d_in[10];
    const float* bg   = (const float*)d_in[11];
    const float* Wf_x = (const float*)d_in[12];
    const float* Wf_n = (const float*)d_in[13];
    const float* Wf_e = (const float*)d_in[14];
    const float* bf   = (const float*)d_in[15];
    const float* Wn   = (const float*)d_in[16];
    const float* bn   = (const float*)d_in[17];
    float* out = (float*)d_out;

    char* ws = (char*)d_ws;
    size_t off = 0;
    auto alloc = [&](size_t bytes) -> void* {
        void* p = ws + off;
        off = (off + bytes + 255) & ~(size_t)255;
        return p;
    };
    const size_t NODE_BYTES = (size_t)N_NODES * EMB * sizeof(float);
    float*    XA   = (float*)alloc(NODE_BYTES);
    float*    XB   = (float*)alloc(NODE_BYTES);
    float*    AG   = (float*)alloc(NODE_BYTES);
    float*    AN   = (float*)alloc(NODE_BYTES);
    float*    AF   = (float*)alloc(NODE_BYTES);
    float*    AFN  = (float*)alloc(NODE_BYTES);
    _Float16* EF16 = (_Float16*)alloc((size_t)N_EDGES * EMB * sizeof(_Float16));
    float*    RBF  = (float*)alloc((size_t)N_EDGES * RBF_DIM * sizeof(float));
    _Float16* W16T = (_Float16*)alloc((size_t)6 * N_LAYERS * EMB * EMB * sizeof(_Float16));
    float*    SUMS = (float*)alloc((size_t)N_GRAPHS * 12 * sizeof(float));
    float*    CNT  = (float*)alloc((size_t)N_GRAPHS * sizeof(float));

    // 1. weights -> transposed f16 (one-time per call, tiny)
    {
        int total = 6 * N_LAYERS * EMB * EMB;
        convert_weights_kernel<<<(total + 255) / 256, 256, 0, stream>>>(
            Wg_x, Wg_n, Wg_e, Wf_x, Wf_n, Wf_e, W16T);
    }
    // 2. embeddings
    embed_kernel<<<(N_NODES * EMB + 255) / 256, 256, 0, stream>>>(x, emb, XA);
    // 3. rbf + edge features (stored f16, feed WMMA directly)
    rbf_kernel<<<(N_EDGES * RBF_DIM + 255) / 256, 256, 0, stream>>>(e, RBF);
    ef_kernel<<<(N_EDGES * EMB + 255) / 256, 256, 0, stream>>>(RBF, We, be, EF16);

    // 4. layers
    float* cur = XA;
    float* nxt = XB;
    for (int l = 0; l < N_LAYERS; ++l) {
        const _Float16* wgx = W16T + (size_t)(0 * N_LAYERS + l) * EMB * EMB;
        const _Float16* wgn = W16T + (size_t)(1 * N_LAYERS + l) * EMB * EMB;
        const _Float16* wge = W16T + (size_t)(2 * N_LAYERS + l) * EMB * EMB;
        const _Float16* wfx = W16T + (size_t)(3 * N_LAYERS + l) * EMB * EMB;
        const _Float16* wfn = W16T + (size_t)(4 * N_LAYERS + l) * EMB * EMB;
        const _Float16* wfe = W16T + (size_t)(5 * N_LAYERS + l) * EMB * EMB;

        node_transform_kernel<<<(N_NODES + 63) / 64, 256, 0, stream>>>(
            cur, wgx, wgn, wfx, wfn, AG, AN, AF, AFN, N_NODES);
        hipMemcpyAsync(nxt, cur, NODE_BYTES, hipMemcpyDeviceToDevice, stream); // residual init
        edge_layer_kernel<<<N_EDGES / 64, 256, 0, stream>>>(
            EF16, AG, AN, AF, AFN, wge, wfe, bg + l * EMB, bf + l * EMB, src, dst, nxt);
        float* t = cur; cur = nxt; nxt = t;
    }

    // 5. readout
    zero_kernel<<<(N_GRAPHS * 12 + 255) / 256, 256, 0, stream>>>(SUMS, CNT);
    readout_kernel<<<(N_NODES * 12 + 255) / 256, 256, 0, stream>>>(cur, Wn, bn, gi, SUMS, CNT);
    finalize_kernel<<<(N_GRAPHS * 12 + 255) / 256, 256, 0, stream>>>(SUMS, CNT, out);
}
